// Attention_29429115912904
// MI455X (gfx1250) — matrix operand order repaired and checked by reading, not compiled, not run on hardware
//
#include <hip/hip_runtime.h>
#include <hip/hip_bf16.h>

// Problem constants (from reference)
#define BB   2
#define SS   2048
#define DD   1024
#define HH   16
#define SPH  64

typedef __attribute__((ext_vector_type(16))) __bf16          bf16x16;
typedef __attribute__((ext_vector_type(16))) unsigned short  u16x16;
typedef __attribute__((ext_vector_type(8)))  float           f32x8;
typedef __attribute__((ext_vector_type(4)))  unsigned int    u32x4;
typedef __attribute__((ext_vector_type(4)))  float           f32x4;

union V16 {
    u16x16 u;
    bf16x16 b;
    u32x4 q[2];
    unsigned short s[16];
};

// Workspace layout (ushort element offsets)
#define OF_XQ  ((size_t)0)                        // query  bf16 [4096,1024]   8 MB
#define OF_XC  ((size_t)1 << 22)                  // context bf16 [4096,1024]  8 MB
#define OF_WT  ((size_t)2 << 22)                  // WqT,WkT,WvT,WoT bf16 [1024,1024] 2 MB each
#define OF_Q   (OF_WT + ((size_t)4 << 20))        // Q  bf16 [B,H,S,64] (pre-scaled by 1/8)
#define OF_K   (OF_Q + ((size_t)1 << 22))         // K  bf16 [B,H,S,64]
#define OF_VT  (OF_K + ((size_t)1 << 22))         // V  bf16 [B,H,64,S] (transposed)
#define OF_AT  (OF_VT + ((size_t)1 << 22))        // attn out bf16 [B,S,D]

__device__ __forceinline__ unsigned short f2bf(float f) {
    unsigned int u = __float_as_uint(f);
    u += 0x7FFFu + ((u >> 16) & 1u);   // round-to-nearest-even
    return (unsigned short)(u >> 16);
}

__device__ __forceinline__ f32x8 wmma_bf16(const V16& a, const V16& b, f32x8 c) {
    return __builtin_amdgcn_wmma_f32_16x16x32_bf16(
        /*neg_a=*/false, a.b, /*neg_b=*/false, b.b,
        /*c_mod=*/(short)0, c, /*reuse_a=*/false, /*reuse_b=*/false);
}

// -----------------------------------------------------------------------------
// Kernel 0a: one-shot fp32 -> bf16 conversion of query / context activations.
// -----------------------------------------------------------------------------
__global__ __launch_bounds__(256) void convert_x_kernel(
    const float* __restrict__ qin, const float* __restrict__ cin,
    unsigned short* __restrict__ ws)
{
    const float* X = blockIdx.z ? cin : qin;
    unsigned short* O = ws + (blockIdx.z ? OF_XC : OF_XQ);
    const size_t idx = ((size_t)blockIdx.x * 256 + threadIdx.x) * 8;
    f32x4 a = *(const f32x4*)(X + idx);
    f32x4 b = *(const f32x4*)(X + idx + 4);
    union { unsigned short s[8]; u32x4 q; } pk;
#pragma unroll
    for (int j = 0; j < 4; ++j) { pk.s[j] = f2bf(a[j]); pk.s[4 + j] = f2bf(b[j]); }
    *(u32x4*)(O + idx) = pk.q;
}

// -----------------------------------------------------------------------------
// Kernel 0b: one-shot weight convert + transpose: W[K,N] fp32 -> WT[N,K] bf16.
// 32x32 tiles through LDS so both global phases are coalesced.
// -----------------------------------------------------------------------------
__global__ __launch_bounds__(256) void convert_wt_kernel(
    const float* __restrict__ Wq, const float* __restrict__ Wk,
    const float* __restrict__ Wv, const float* __restrict__ Wo,
    unsigned short* __restrict__ ws)
{
    const int m = blockIdx.z;
    const float* W = (m == 0) ? Wq : (m == 1) ? Wk : (m == 2) ? Wv : Wo;
    unsigned short* O = ws + OF_WT + (size_t)m * ((size_t)1 << 20);
    const int k0 = blockIdx.x * 32;
    const int n0 = blockIdx.y * 32;
    __shared__ unsigned short Lt[32][36];   // [n][k] with pad
    const int tid = threadIdx.x;
    {
        const int row = tid >> 3;           // k local
        const int c4  = (tid & 7) * 4;      // n local
        f32x4 w = *(const f32x4*)(W + (size_t)(k0 + row) * DD + n0 + c4);
#pragma unroll
        for (int j = 0; j < 4; ++j) Lt[c4 + j][row] = f2bf(w[j]);
    }
    __syncthreads();
    {
        const int n  = tid >> 3;
        const int k4 = (tid & 7) * 4;
        union { unsigned short s[4]; unsigned long long q; } pk;
#pragma unroll
        for (int j = 0; j < 4; ++j) pk.s[j] = Lt[n][k4 + j];
        *(unsigned long long*)(O + (size_t)(n0 + n) * DD + k0 + k4) = pk.q;
    }
}

// -----------------------------------------------------------------------------
// Kernel 1: QKV projection GEMMs, all-register, no LDS.
//  A: Xbf [4096,1024] bf16 row-major (direct A-layout loads)
//  B: WT  [1024 n,1024 k] bf16 (contiguous 16B loads per lane)
//  sel 0 -> Q [B,H,S,64] scaled by 1/8 ; sel 1 -> K [B,H,S,64] ; sel 2 -> V^T [B,H,64,S]
// -----------------------------------------------------------------------------
__global__ __launch_bounds__(256) void qkv_gemm_kernel(unsigned short* __restrict__ ws)
{
    const int sel = blockIdx.z;
    const unsigned short* X  = ws + (sel == 0 ? OF_XQ : OF_XC);
    const unsigned short* WT = ws + OF_WT + (size_t)sel * ((size_t)1 << 20);
    unsigned short* out = ws + OF_Q + (size_t)sel * ((size_t)1 << 22);
    const int head = blockIdx.y;
    const int m0   = blockIdx.x * 128;
    const int b    = m0 / SS;
    const int s0   = m0 - b * SS;

    const int tid  = threadIdx.x;
    const int wave = tid >> 5;
    const int lane = tid & 31;
    const int lh   = lane & 15;
    const int hi   = lane >> 4;

    f32x8 acc[4] = {};
    const unsigned short* Arow = X + (size_t)(m0 + wave * 16 + lh) * DD;
    const int koA = hi ? 16 : 0;
    const unsigned short* Bbase = WT + (size_t)(head * SPH + lh) * DD + hi * 16;

#pragma unroll 2
    for (int k0 = 0; k0 < DD; k0 += 32) {
        V16 a;
        a.q[0] = *(const u32x4*)(Arow + k0 + koA);
        a.q[1] = *(const u32x4*)(Arow + k0 + koA + 8);
        V16 bb[4];
#pragma unroll
        for (int g = 0; g < 4; ++g) {
            const unsigned short* bp = Bbase + (size_t)(g * 16) * DD + k0;
            bb[g].q[0] = *(const u32x4*)bp;
            bb[g].q[1] = *(const u32x4*)(bp + 8);
        }
#pragma unroll
        for (int g = 0; g < 4; ++g) acc[g] = wmma_bf16(a, bb[g], acc[g]);
    }

    if (sel < 2) {
        const float oscale = (sel == 0) ? 0.125f : 1.0f;   // fold 1/sqrt(SPH) into Q
        unsigned short* ob = out + ((size_t)(b * HH + head) * SS) * SPH;
        const int sb = s0 + wave * 16 + hi * 8;
#pragma unroll
        for (int g = 0; g < 4; ++g) {
            const int n = g * 16 + lh;
#pragma unroll
            for (int i = 0; i < 8; ++i)
                ob[(size_t)(sb + i) * SPH + n] = f2bf(acc[g][i] * oscale);
        }
    } else {
        unsigned short* ob = out + ((size_t)(b * HH + head) * SPH) * SS;
        const int sb = s0 + wave * 16 + hi * 8;
#pragma unroll
        for (int g = 0; g < 4; ++g) {
            const int n = g * 16 + lh;
            union { unsigned short s[8]; u32x4 q; } pk;
#pragma unroll
            for (int i = 0; i < 8; ++i) pk.s[i] = f2bf(acc[g][i]);
            *(u32x4*)&ob[(size_t)n * SS + sb] = pk.q;
        }
    }
}

// -----------------------------------------------------------------------------
// Kernel 2: flash attention with online softmax.
// grid (S/128, H, B), 8 waves/block; each wave: 16-row Q tile, context streamed
// in 32-wide KV tiles. Q pre-scaled; P routed C-layout -> A-layout through LDS.
// -----------------------------------------------------------------------------
__global__ __launch_bounds__(256) void flash_attn_kernel(
    const unsigned short* __restrict__ ws, const float* __restrict__ mask,
    unsigned short* __restrict__ attn)
{
    const int b    = blockIdx.z;
    const int head = blockIdx.y;
    const int s0   = blockIdx.x * 128;
    const int wave = threadIdx.x >> 5;
    const int lane = threadIdx.x & 31;
    const int lh   = lane & 15;
    const int hi   = lane >> 4;

    const unsigned short* qb = ws + OF_Q  + (size_t)(b * HH + head) * SS * SPH;
    const unsigned short* kb = ws + OF_K  + (size_t)(b * HH + head) * SS * SPH;
    const unsigned short* vT = ws + OF_VT + (size_t)(b * HH + head) * SPH * SS;

    __shared__ unsigned short P[8][16 * 32];   // per-wave P staging

    V16 aq[2];
    {
        const unsigned short* qr = qb + (size_t)(s0 + wave * 16 + lh) * SPH;
        const int ko = hi ? 16 : 0;
#pragma unroll
        for (int t = 0; t < 2; ++t) {
            aq[t].q[0] = *(const u32x4*)(qr + t * 32 + ko);
            aq[t].q[1] = *(const u32x4*)(qr + t * 32 + ko + 8);
        }
    }

    f32x8 acc[4] = {};
    float mrow[8], lrow[8];
#pragma unroll
    for (int i = 0; i < 8; ++i) { mrow[i] = -1e30f; lrow[i] = 0.0f; }

    const float* mb = mask + (size_t)b * SS * SS;
    const int rbase = s0 + wave * 16 + hi * 8;

    for (int c0 = 0; c0 < SS; c0 += 32) {
        // prefetch next KV tile while we crunch this one
        if (c0 + 32 < SS) {
            __builtin_prefetch(kb + (size_t)(c0 + 32 + lane) * SPH, 0, 3);
            __builtin_prefetch(vT + (size_t)lane * SS + c0 + 32, 0, 3);
            __builtin_prefetch(vT + (size_t)(32 + lane) * SS + c0 + 32, 0, 3);
        }

        // batch-load all K B-operands, then issue score WMMAs
        V16 bk[2][2];
#pragma unroll
        for (int j = 0; j < 2; ++j)
#pragma unroll
            for (int t = 0; t < 2; ++t) {
                const unsigned short* kr =
                    kb + (size_t)(c0 + j * 16 + lh) * SPH + t * 32 + hi * 16;
                bk[j][t].q[0] = *(const u32x4*)kr;
                bk[j][t].q[1] = *(const u32x4*)(kr + 8);
            }
        f32x8 sc[2];
#pragma unroll
        for (int j = 0; j < 2; ++j) {
            f32x8 t0 = {};
            t0 = wmma_bf16(aq[0], bk[j][0], t0);
            t0 = wmma_bf16(aq[1], bk[j][1], t0);
            sc[j] = t0;
        }

        // online softmax (Q already carries the 1/8 scale)
#pragma unroll
        for (int i = 0; i < 8; ++i) {
            const int r = rbase + i;
            float x0 = fmaf(mb[(size_t)r * SS + c0 + lh],      -1e9f, sc[0][i]);
            float x1 = fmaf(mb[(size_t)r * SS + c0 + 16 + lh], -1e9f, sc[1][i]);
            float mx = fmaxf(x0, x1);
            mx = fmaxf(mx, __shfl_xor(mx, 1, 32));
            mx = fmaxf(mx, __shfl_xor(mx, 2, 32));
            mx = fmaxf(mx, __shfl_xor(mx, 4, 32));
            mx = fmaxf(mx, __shfl_xor(mx, 8, 32));
            const float mnew = fmaxf(mrow[i], mx);
            const float al   = __expf(mrow[i] - mnew);
            const float p0   = __expf(x0 - mnew);
            const float p1   = __expf(x1 - mnew);
            float rs = p0 + p1;
            rs += __shfl_xor(rs, 1, 32);
            rs += __shfl_xor(rs, 2, 32);
            rs += __shfl_xor(rs, 4, 32);
            rs += __shfl_xor(rs, 8, 32);
            lrow[i] = lrow[i] * al + rs;
            mrow[i] = mnew;
#pragma unroll
            for (int g = 0; g < 4; ++g) acc[g][i] *= al;
            const int M = i + hi * 8;
            P[wave][M * 32 + lh]      = f2bf(p0);
            P[wave][M * 32 + 16 + lh] = f2bf(p1);
        }
        __builtin_amdgcn_wave_barrier();   // same-wave LDS ops are in-order (ISA 7.3)

        V16 ap;
        {
            const unsigned short* pr = &P[wave][lh * 32 + hi * 16];
            ap.q[0] = *(const u32x4*)pr;
            ap.q[1] = *(const u32x4*)(pr + 8);
        }
        __builtin_amdgcn_wave_barrier();

        // batch-load all V B-operands, then PV WMMAs
        V16 bv[4];
#pragma unroll
        for (int g = 0; g < 4; ++g) {
            const unsigned short* vr = vT + (size_t)(g * 16 + lh) * SS + c0 + hi * 16;
            bv[g].q[0] = *(const u32x4*)vr;
            bv[g].q[1] = *(const u32x4*)(vr + 8);
        }
#pragma unroll
        for (int g = 0; g < 4; ++g) acc[g] = wmma_bf16(ap, bv[g], acc[g]);
    }

    // normalize and store to [B,S,D] bf16
#pragma unroll
    for (int g = 0; g < 4; ++g) {
        const int col = head * SPH + g * 16 + lh;
#pragma unroll
        for (int i = 0; i < 8; ++i) {
            const int r = rbase + i;
            attn[(size_t)(b * SS + r) * DD + col] = f2bf(acc[g][i] / lrow[i]);
        }
    }
}

// -----------------------------------------------------------------------------
// Kernel 3: output projection, all-register. attn bf16 x WoT bf16 -> fp32 out.
// -----------------------------------------------------------------------------
__global__ __launch_bounds__(256) void out_proj_kernel(
    const unsigned short* __restrict__ ws_ro, float* __restrict__ out)
{
    const unsigned short* attn = ws_ro + OF_AT;
    const unsigned short* WoT  = ws_ro + OF_WT + ((size_t)3 << 20);
    const int m0 = blockIdx.x * 128;
    const int n0 = blockIdx.y * 64;
    const int tid  = threadIdx.x;
    const int wave = tid >> 5;
    const int lane = tid & 31;
    const int lh   = lane & 15;
    const int hi   = lane >> 4;

    f32x8 acc[4] = {};
    const unsigned short* Arow  = attn + (size_t)(m0 + wave * 16 + lh) * DD;
    const int koA = hi ? 16 : 0;
    const unsigned short* Bbase = WoT + (size_t)(n0 + lh) * DD + hi * 16;

#pragma unroll 2
    for (int k0 = 0; k0 < DD; k0 += 32) {
        V16 a;
        a.q[0] = *(const u32x4*)(Arow + k0 + koA);
        a.q[1] = *(const u32x4*)(Arow + k0 + koA + 8);
        V16 bb[4];
#pragma unroll
        for (int g = 0; g < 4; ++g) {
            const unsigned short* bp = Bbase + (size_t)(g * 16) * DD + k0;
            bb[g].q[0] = *(const u32x4*)bp;
            bb[g].q[1] = *(const u32x4*)(bp + 8);
        }
#pragma unroll
        for (int g = 0; g < 4; ++g) acc[g] = wmma_bf16(a, bb[g], acc[g]);
    }

    const int rb = m0 + wave * 16 + hi * 8;
#pragma unroll
    for (int g = 0; g < 4; ++g) {
        const int col = n0 + g * 16 + lh;
#pragma unroll
        for (int i = 0; i < 8; ++i)
            out[(size_t)(rb + i) * DD + col] = acc[g][i];
    }
}

// -----------------------------------------------------------------------------
extern "C" void kernel_launch(void* const* d_in, const int* in_sizes, int n_in,
                              void* d_out, int out_size, void* d_ws, size_t ws_size,
                              hipStream_t stream) {
    const float* query   = (const float*)d_in[0];
    const float* context = (const float*)d_in[1];
    const float* mask    = (const float*)d_in[2];
    const float* Wq      = (const float*)d_in[3];
    const float* Wk      = (const float*)d_in[4];
    const float* Wv      = (const float*)d_in[5];
    const float* Wo      = (const float*)d_in[6];
    unsigned short* ws   = (unsigned short*)d_ws;

    convert_x_kernel <<<dim3(2048, 1, 2),  256, 0, stream>>>(query, context, ws);
    convert_wt_kernel<<<dim3(32, 32, 4),   256, 0, stream>>>(Wq, Wk, Wv, Wo, ws);
    qkv_gemm_kernel  <<<dim3(32, 16, 3),   256, 0, stream>>>(ws);
    flash_attn_kernel<<<dim3(16, 16, 2),   256, 0, stream>>>(ws, mask, ws + OF_AT);
    out_proj_kernel  <<<dim3(32, 16, 1),   256, 0, stream>>>(ws, (float*)d_out);
}
